// LocalAttentionBlock_75445395521923
// MI455X (gfx1250) — matrix operand order repaired and checked by reading, not compiled, not run on hardware
//
#include <hip/hip_runtime.h>
#include <hip/hip_bf16.h>
#include <math.h>

#define T_   4096
#define D_   1024
#define H_   16
#define HD_  64
#define WIN_ 128
#define MLP_ 4096

// GEMM blocking: 256 threads (8 waves), block tile 128x128, K-panel 64,
// wave tile 64x32 (4x2 sub-tiles of 16x16), double-buffered LDS panels.
#define BM_  128
#define BN_  128
#define BK_  64
#define PAD_ 8                     // ushort pad per LDS row (bank spread)
#define LROW_ (BK_ + PAD_)         // 72 ushorts = 144 B row stride

typedef __attribute__((ext_vector_type(16))) __bf16 v16bf;
typedef __attribute__((ext_vector_type(8)))  float  v8f;
typedef __attribute__((ext_vector_type(4)))  unsigned v4u;
typedef __attribute__((ext_vector_type(8)))  int      v8i;
typedef __attribute__((ext_vector_type(4)))  int      v4i;

union FragB16 { unsigned u[8]; v16bf v; };

// native RNE f32 -> bf16 (v_cvt_*bf16_f32)
__device__ __forceinline__ unsigned short f2bf(float f) {
  __bf16 h = (__bf16)f;
  return __builtin_bit_cast(unsigned short, h);
}
__device__ __forceinline__ unsigned pack2bf(float lo, float hi) {
  return (unsigned)f2bf(lo) | ((unsigned)f2bf(hi) << 16);
}

// ---- CDNA5 async global->LDS copy (ASYNCcnt) ---------------------------
__device__ __forceinline__ void async_cp_b128(unsigned lds_off, const unsigned short* g) {
  asm volatile("global_load_async_to_lds_b128 %0, %1, off"
               :: "v"(lds_off), "v"(g) : "memory");
}
__device__ __forceinline__ void wait_async0() {
  asm volatile("s_wait_asynccnt 0x0" ::: "memory");
}

// ---- CDNA5 Tensor Data Mover: 2D bf16 tile global->LDS (TENSORcnt) -----
// Builds a D# per ISA ch.8: group0 = count/lds_addr/global_addr/type,
// group1 = data_size + LDS row padding + tensor/tile dims + dim0 stride.
// pad_interval code 4 = every 32 DWORDs (128B = one 64-elem bf16 row),
// pad_amount  code 3 = 4 DWORDs (16B) -> reproduces the 144B LROW_ stride.
__device__ __forceinline__ void tdm_load_2d(unsigned lds_addr, const void* gptr,
                                            unsigned tensor_d0, unsigned tensor_d1,
                                            unsigned tile_d0, unsigned tile_d1,
                                            unsigned stride0) {
  const unsigned long long ga = (unsigned long long)gptr;
  v4u g0;
  g0[0] = 1u;                                            // count=1, user D#
  g0[1] = lds_addr;                                      // LDS byte address
  g0[2] = (unsigned)(ga & 0xFFFFFFFFull);                // global_addr[31:0]
  g0[3] = (unsigned)((ga >> 32) & 0x1FFFFFFull)          // global_addr[56:32]
        | (2u << 30);                                    // type = 2 ("image")
  v8i g1;
  g1[0] = (int)((1u << 16)                               // data_size = 2 bytes
        | (1u << 20) | (4u << 22) | (3u << 25));         // pad_en, interval, amount
  g1[1] = (int)((tensor_d0 & 0xFFFFu) << 16);            // tensor_dim0[15:0]
  g1[2] = (int)(((tensor_d0 >> 16) & 0xFFFFu)            // tensor_dim0[31:16]
        | ((tensor_d1 & 0xFFFFu) << 16));                // tensor_dim1[15:0]
  g1[3] = (int)(((tensor_d1 >> 16) & 0xFFFFu)            // tensor_dim1[31:16]
        | ((tile_d0 & 0xFFFFu) << 16));                  // tile_dim0
  g1[4] = (int)(tile_d1 & 0xFFFFu);                      // tile_dim1 (tile_dim2=0)
  g1[5] = (int)stride0;                                  // dim0_stride[31:0]
  g1[6] = 0;                                             // stride hi + dim1_stride
  g1[7] = 0;
  const v4i gz  = {0, 0, 0, 0};                          // 2D: groups 2/3 unused
  const v8i gz8 = {0, 0, 0, 0, 0, 0, 0, 0};              // 6-arg toolchain extra group
  __builtin_amdgcn_tensor_load_to_lds(g0, g1, gz, gz, gz8, 0);
}
__device__ __forceinline__ void wait_tensor0() {
  __builtin_amdgcn_s_wait_tensorcnt(0);
}

// ---- VALU DPP16 row (16-lane) max reduction: no LDS, no dscnt waits ----
template <int CTRL>
__device__ __forceinline__ float dpp_mov_f(float x) {
  return __builtin_bit_cast(float,
      __builtin_amdgcn_mov_dpp(__builtin_bit_cast(int, x), CTRL, 0xf, 0xf, true));
}
__device__ __forceinline__ float rowmax16(float x) {
  x = fmaxf(x, dpp_mov_f<0xB1>(x));    // quad_perm(1,0,3,2)  : xor 1
  x = fmaxf(x, dpp_mov_f<0x4E>(x));    // quad_perm(2,3,0,1)  : xor 2
  x = fmaxf(x, dpp_mov_f<0x141>(x));   // row_half_mirror     : xor 7 (4-groups uniform)
  x = fmaxf(x, dpp_mov_f<0x140>(x));   // row_mirror          : xor 15
  return x;
}

// ---- WMMA fragment helpers (wave32, bf16, 16x16x32) --------------------
__device__ __forceinline__ void load_a_frag(const unsigned short* src, int ld,
                                            int row0, int k0, unsigned* u) {
  const int lane = threadIdx.x & 31;
  const int m = lane & 15, half = lane >> 4;
  const unsigned short* p = src + (size_t)(row0 + m) * ld + k0 + half * 8;
  *(uint4*)(u)     = *(const uint4*)(p);
  *(uint4*)(u + 4) = *(const uint4*)(p + 16);
}
template <int STRIDE>
__device__ __forceinline__ void lds_b_frag(const unsigned short* base, int r0,
                                           int k0, unsigned* u) {
  const int lane = threadIdx.x & 31;
  const int n = lane & 15, half = lane >> 4;
  const unsigned short* p = base + (r0 + n) * STRIDE + k0 + half * 8;
  *(uint4*)(u)     = *(const uint4*)(p);
  *(uint4*)(u + 4) = *(const uint4*)(p + 16);
}

__device__ __forceinline__ v8f wmma_bf16(const FragB16& a, const FragB16& b, v8f c) {
  return __builtin_amdgcn_wmma_f32_16x16x32_bf16(false, a.v, false, b.v,
                                                 (short)0, c, false, false);
}

// ---- elementwise / normalization kernels -------------------------------
__global__ void k_cvt_bf16(const float* __restrict__ in,
                           unsigned short* __restrict__ out, int n) {
  const int n8 = n >> 3;
  const float4* in4 = (const float4*)in;
  uint4* out4 = (uint4*)out;
  int i = blockIdx.x * blockDim.x + threadIdx.x;
  const int stride = gridDim.x * blockDim.x;
  for (; i < n8; i += stride) {
    const float4 a = in4[i * 2], b = in4[i * 2 + 1];
    uint4 o;
    o.x = pack2bf(a.x, a.y);
    o.y = pack2bf(a.z, a.w);
    o.z = pack2bf(b.x, b.y);
    o.w = pack2bf(b.z, b.w);
    out4[i] = o;
  }
}

__global__ void k_layernorm(const float* __restrict__ x, const float* __restrict__ g,
                            const float* __restrict__ b, unsigned short* __restrict__ out) {
  __shared__ float red[256];
  const int row = blockIdx.x, tid = threadIdx.x;
  const float4 v4 = ((const float4*)(x + (size_t)row * D_))[tid];
  red[tid] = v4.x + v4.y + v4.z + v4.w;
  __syncthreads();
  for (int w = 128; w > 0; w >>= 1) {
    if (tid < w) red[tid] += red[tid + w];
    __syncthreads();
  }
  const float mean = red[0] * (1.f / D_);
  __syncthreads();
  const float d0 = v4.x - mean, d1 = v4.y - mean, d2 = v4.z - mean, d3 = v4.w - mean;
  red[tid] = d0 * d0 + d1 * d1 + d2 * d2 + d3 * d3;
  __syncthreads();
  for (int w = 128; w > 0; w >>= 1) {
    if (tid < w) red[tid] += red[tid + w];
    __syncthreads();
  }
  const float rstd = rsqrtf(red[0] * (1.f / D_) + 1e-5f);
  const float4 g4 = ((const float4*)g)[tid];
  const float4 b4 = ((const float4*)b)[tid];
  uint2 o;
  o.x = pack2bf(d0 * rstd * g4.x + b4.x, d1 * rstd * g4.y + b4.y);
  o.y = pack2bf(d2 * rstd * g4.z + b4.z, d3 * rstd * g4.w + b4.w);
  ((uint2*)(out + (size_t)row * D_))[tid] = o;
}

// ---- blocked GEMM with TDM panel loads ---------------------------------
// MODE 0: QKV (bias, scatter to head-major q/k/v bf16)
// MODE 1: out-proj (bias + residual -> f32)
// MODE 2: MLP up (bias + exact GELU -> bf16, ld MLP_)
// MODE 3: MLP down (bias + residual -> f32 final)
template <int MODE>
__global__ void k_gemm(const unsigned short* __restrict__ A,
                       const unsigned short* __restrict__ W, int K,
                       const float* __restrict__ bias,
                       const float* __restrict__ resid,
                       float* __restrict__ outf,
                       unsigned short* __restrict__ outh,
                       unsigned short* __restrict__ oq,
                       unsigned short* __restrict__ ok,
                       unsigned short* __restrict__ ov) {
  __shared__ __attribute__((aligned(16))) unsigned short sA[2][BM_][LROW_];
  __shared__ __attribute__((aligned(16))) unsigned short sB[2][BN_][LROW_];

  const int tid   = threadIdx.x;
  const int lane  = tid & 31;
  const int wid   = tid >> 5;            // 0..7
  const int wm    = wid & 1;             // 2 wave rows  (64 M each)
  const int wn    = wid >> 1;            // 4 wave cols  (32 N each)
  const int m0    = blockIdx.y * BM_;
  const int n0    = blockIdx.x * BN_;
  const int mlane = lane & 15, half = lane >> 4;

  auto issue = [&](int buf, int k0) {    // wave 0 issues both panel DMAs
    if (wid == 0) {
      tdm_load_2d((unsigned)(size_t)&sA[buf][0][0],
                  A + (size_t)m0 * K + k0, K, BM_ * 32, BK_, BM_, K);
      tdm_load_2d((unsigned)(size_t)&sB[buf][0][0],
                  W + (size_t)n0 * K + k0, K, BN_ * 32, BK_, BN_, K);
    }
  };

  v8f acc[4][2];
  const v8f z = {};
#pragma unroll
  for (int i = 0; i < 4; ++i)
#pragma unroll
    for (int j = 0; j < 2; ++j) acc[i][j] = z;

  issue(0, 0);
  int buf = 0;
  for (int k0 = 0; k0 < K; k0 += BK_) {
    if (wid == 0) wait_tensor0();        // this panel's two TDM ops complete
    __syncthreads();                     // panels visible to all waves
    if (k0 + BK_ < K) issue(buf ^ 1, k0 + BK_);   // prefetch next panel
#pragma unroll
    for (int ks = 0; ks < BK_; ks += 32) {
      FragB16 af[4], bf2[2];
#pragma unroll
      for (int i = 0; i < 4; ++i) {
        const unsigned short* pr = &sA[buf][wm * 64 + i * 16 + mlane][ks + half * 8];
        *(uint4*)(af[i].u)     = *(const uint4*)(pr);
        *(uint4*)(af[i].u + 4) = *(const uint4*)(pr + 16);
      }
#pragma unroll
      for (int j = 0; j < 2; ++j) {
        const unsigned short* pr = &sB[buf][wn * 32 + j * 16 + mlane][ks + half * 8];
        *(uint4*)(bf2[j].u)     = *(const uint4*)(pr);
        *(uint4*)(bf2[j].u + 4) = *(const uint4*)(pr + 16);
      }
#pragma unroll
      for (int i = 0; i < 4; ++i)
#pragma unroll
        for (int j = 0; j < 2; ++j)
          acc[i][j] = wmma_bf16(af[i], bf2[j], acc[i][j]);
    }
    __syncthreads();                     // done reading buf before next overwrite
    buf ^= 1;
  }

#pragma unroll
  for (int i = 0; i < 4; ++i)
#pragma unroll
    for (int j = 0; j < 2; ++j)
#pragma unroll
      for (int r = 0; r < 8; ++r) {
        const int row = m0 + wm * 64 + i * 16 + r + 8 * half;
        const int col = n0 + wn * 32 + j * 16 + mlane;
        const float v = acc[i][j][r];
        if constexpr (MODE == 0) {
          const float val = v + bias[col];
          const int sel = col >> 10, rem = col & 1023;
          const int head = rem >> 6, d = rem & 63;
          unsigned short* dst = (sel == 0) ? oq : (sel == 1) ? ok : ov;
          dst[((size_t)head * T_ + row) * HD_ + d] = f2bf(val);
        } else if constexpr (MODE == 2) {
          float val = v + bias[col];
          val = 0.5f * val * (1.f + erff(val * 0.70710678118654752f));
          outh[(size_t)row * MLP_ + col] = f2bf(val);
        } else {
          const size_t idx = (size_t)row * D_ + col;
          outf[idx] = resid[idx] + v + bias[col];
        }
      }
}

// ---- windowed flash attention --------------------------------------------
// Block = 4 waves = 4 consecutive query tiles (64 rows) of one head.
// Shared union key window: 20 tiles -> 10 uniform iterations of 32 keys;
// K tiles async-copied to LDS, V tiles transposed into LDS.
// Row-max via DPP16 butterflies; row-sum via P @ ones WMMA.
__global__ void k_attn(const unsigned short* __restrict__ q,
                       const unsigned short* __restrict__ kg,
                       const unsigned short* __restrict__ vg,
                       unsigned short* __restrict__ o) {
  __shared__ __attribute__((aligned(16))) unsigned short sK[32][72];     // keys x dims
  __shared__ __attribute__((aligned(16))) unsigned short sVt[64][40];    // dims x keys
  __shared__ __attribute__((aligned(16))) unsigned short sPb[4][16][40]; // per-wave P (bf16)

  const int tid  = threadIdx.x;
  const int wv   = tid >> 5;
  const int lane = tid & 31;
  const int nidx = lane & 15, half = lane >> 4;
  const int qb   = blockIdx.x * 64;      // query block base
  const int m0   = qb + wv * 16;         // this wave's query tile
  const int h    = blockIdx.y;

  const unsigned short* Qh = q  + (size_t)h * T_ * HD_;
  const unsigned short* Kh = kg + (size_t)h * T_ * HD_;
  const unsigned short* Vh = vg + (size_t)h * T_ * HD_;

  FragB16 qa0, qa1;
  load_a_frag(Qh, HD_, m0, 0,  qa0.u);
  load_a_frag(Qh, HD_, m0, 32, qa1.u);

  FragB16 onesf;                          // all-ones B fragment (bf16 1.0)
#pragma unroll
  for (int i = 0; i < 8; ++i) onesf.u[i] = 0x3F803F80u;

  v8f accO[4];
  const v8f z = {};
#pragma unroll
  for (int c = 0; c < 4; ++c) accO[c] = z;
  float rm[8], rl[8];
#pragma unroll
  for (int r = 0; r < 8; ++r) { rm[r] = -3.0e38f; rl[r] = 0.f; }

  const int   tu0   = (qb >> 4) - 8;     // union window start tile
  const float scale = 0.125f;            // 1/sqrt(HD)

  for (int step = 0; step < 10; ++step) {
    const int key0 = (tu0 + step * 2) * 16;   // logical first key (may be < 0)

    __syncthreads();                     // prior reads of sK/sVt complete
#pragma unroll
    for (int p = 0; p < 2; ++p) {
      const int c = tid + p * 128;       // 256 x 16B chunks
      const int krow = c >> 3, c16 = c & 7;
      int grow = key0 + krow; if (grow < 0) grow = 0; if (grow > T_ - 1) grow = T_ - 1;
      async_cp_b128((unsigned)(size_t)&sK[krow][c16 * 8],
                    Kh + (size_t)grow * HD_ + c16 * 8);
      const uint4 w = *(const uint4*)(Vh + (size_t)grow * HD_ + c16 * 8);
      const unsigned wd[4] = {w.x, w.y, w.z, w.w};
#pragma unroll
      for (int jj = 0; jj < 4; ++jj) {
        sVt[c16 * 8 + 2 * jj + 0][krow] = (unsigned short)(wd[jj] & 0xFFFFu);
        sVt[c16 * 8 + 2 * jj + 1][krow] = (unsigned short)(wd[jj] >> 16);
      }
    }
    wait_async0();
    __syncthreads();                     // K + V tiles visible to all waves

    FragB16 kb00, kb01, kb10, kb11;
    lds_b_frag<72>(&sK[0][0], 0,  0,  kb00.u);
    lds_b_frag<72>(&sK[0][0], 0,  32, kb01.u);
    lds_b_frag<72>(&sK[0][0], 16, 0,  kb10.u);
    lds_b_frag<72>(&sK[0][0], 16, 32, kb11.u);
    v8f s0 = z, s1 = z;
    s0 = wmma_bf16(qa0, kb00, s0);
    s1 = wmma_bf16(qa0, kb10, s1);
    s0 = wmma_bf16(qa1, kb01, s0);
    s1 = wmma_bf16(qa1, kb11, s1);

#pragma unroll
    for (int r = 0; r < 8; ++r) {
      const int row = m0 + r + 8 * half;
      const int c0 = key0 + nidx, c1 = key0 + 16 + nidx;
      float sv0 = s0[r] * scale;
      float sv1 = s1[r] * scale;
      { int d = row - c0; if (d < 0) d = -d;
        if (d > WIN_ || c0 < 0 || c0 >= T_) sv0 = -3.0e38f; }
      { int d = row - c1; if (d < 0) d = -d;
        if (d > WIN_ || c1 < 0 || c1 >= T_) sv1 = -3.0e38f; }
      const float mx   = rowmax16(fmaxf(sv0, sv1));     // DPP, no LDS
      const float mnew = fmaxf(rm[r], mx);
      const float alpha = expf(rm[r] - mnew);
      const float e0 = (sv0 < -1.0e38f) ? 0.f : expf(sv0 - mnew);
      const float e1 = (sv1 < -1.0e38f) ? 0.f : expf(sv1 - mnew);
      rm[r] = mnew;
      rl[r] *= alpha;
#pragma unroll
      for (int c = 0; c < 4; ++c) accO[c][r] *= alpha;
      sPb[wv][r + 8 * half][nidx]      = f2bf(e0);
      sPb[wv][r + 8 * half][16 + nidx] = f2bf(e1);
    }
    __builtin_amdgcn_wave_barrier();     // DS in-order within wave
    FragB16 pa;
#pragma unroll
    for (int vv = 0; vv < 8; ++vv) {
      const int kk = ((vv >> 2) << 4) + (half << 3) + ((vv & 3) << 1);
      pa.u[vv] = *(const unsigned*)&sPb[wv][nidx][kk];
    }

    const v8f rsv = wmma_bf16(pa, onesf, z);            // row sums via WMMA
#pragma unroll
    for (int r = 0; r < 8; ++r) rl[r] += rsv[r];

    FragB16 vb[4];
#pragma unroll
    for (int c = 0; c < 4; ++c)
      lds_b_frag<40>(&sVt[0][0], c * 16, 0, vb[c].u);
#pragma unroll
    for (int c = 0; c < 4; ++c)
      accO[c] = wmma_bf16(pa, vb[c], accO[c]);
  }

#pragma unroll
  for (int r = 0; r < 8; ++r) {
    const int row = m0 + r + 8 * half;
    const float inv = 1.f / rl[r];
#pragma unroll
    for (int c = 0; c < 4; ++c)
      o[(size_t)row * D_ + h * HD_ + c * 16 + nidx] = f2bf(accO[c][r] * inv);
  }
}

// ---- host orchestration ------------------------------------------------
extern "C" void kernel_launch(void* const* d_in, const int* in_sizes, int n_in,
                              void* d_out, int out_size, void* d_ws, size_t ws_size,
                              hipStream_t stream) {
  (void)in_sizes; (void)n_in; (void)out_size; (void)ws_size;
  const float* x     = (const float*)d_in[0];
  const float* w_qkv = (const float*)d_in[1];
  const float* b_qkv = (const float*)d_in[2];
  const float* w_out = (const float*)d_in[3];
  const float* b_out = (const float*)d_in[4];
  const float* g1    = (const float*)d_in[5];
  const float* be1   = (const float*)d_in[6];
  const float* g2    = (const float*)d_in[7];
  const float* be2   = (const float*)d_in[8];
  const float* w1    = (const float*)d_in[9];
  const float* b1    = (const float*)d_in[10];
  const float* w2    = (const float*)d_in[11];
  const float* b2    = (const float*)d_in[12];
  float* out = (float*)d_out;

  char* ws = (char*)d_ws;
  size_t off = 0;
  auto carve = [&](size_t bytes) -> char* {
    char* p = ws + off; off += (bytes + 255) & ~(size_t)255; return p;
  };
  unsigned short* xn    = (unsigned short*)carve((size_t)T_ * D_ * 2);
  unsigned short* wqkvb = (unsigned short*)carve((size_t)3 * D_ * D_ * 2);
  unsigned short* woutb = (unsigned short*)carve((size_t)D_ * D_ * 2);
  unsigned short* w1b   = (unsigned short*)carve((size_t)MLP_ * D_ * 2);
  unsigned short* w2b   = (unsigned short*)carve((size_t)D_ * MLP_ * 2);
  unsigned short* qh    = (unsigned short*)carve((size_t)T_ * D_ * 2);
  unsigned short* kh    = (unsigned short*)carve((size_t)T_ * D_ * 2);
  unsigned short* vh    = (unsigned short*)carve((size_t)T_ * D_ * 2);
  unsigned short* ob    = (unsigned short*)carve((size_t)T_ * D_ * 2);
  float*          x1    = (float*)carve((size_t)T_ * D_ * 4);
  unsigned short* xn2   = (unsigned short*)carve((size_t)T_ * D_ * 2);
  unsigned short* hbuf  = (unsigned short*)carve((size_t)T_ * MLP_ * 2);

  k_cvt_bf16<<<2048, 256, 0, stream>>>(w_qkv, wqkvb, 3 * D_ * D_);
  k_cvt_bf16<<<2048, 256, 0, stream>>>(w_out, woutb, D_ * D_);
  k_cvt_bf16<<<2048, 256, 0, stream>>>(w1,    w1b,   MLP_ * D_);
  k_cvt_bf16<<<2048, 256, 0, stream>>>(w2,    w2b,   D_ * MLP_);

  k_layernorm<<<T_, 256, 0, stream>>>(x, g1, be1, xn);
  k_gemm<0><<<dim3(3 * D_ / BN_, T_ / BM_), 256, 0, stream>>>(
      xn, wqkvb, D_, b_qkv, nullptr, nullptr, nullptr, qh, kh, vh);
  k_attn<<<dim3(T_ / 64, H_), 128, 0, stream>>>(qh, kh, vh, ob);
  k_gemm<1><<<dim3(D_ / BN_, T_ / BM_), 256, 0, stream>>>(
      ob, woutb, D_, b_out, x, x1, nullptr, nullptr, nullptr, nullptr);
  k_layernorm<<<T_, 256, 0, stream>>>(x1, g2, be2, xn2);
  k_gemm<2><<<dim3(MLP_ / BN_, T_ / BM_), 256, 0, stream>>>(
      xn2, w1b, D_, b1, nullptr, nullptr, hbuf, nullptr, nullptr, nullptr);
  k_gemm<3><<<dim3(D_ / BN_, T_ / BM_), 256, 0, stream>>>(
      hbuf, w2b, MLP_, b2, x1, out, nullptr, nullptr, nullptr, nullptr);
}
